// MultiHeadAttention_61641370632371
// MI455X (gfx1250) — matrix operand order repaired and checked by reading, not compiled, not run on hardware
//
#include <hip/hip_runtime.h>

// ---------------------------------------------------------------------------
// MHA B=2,S=2048,D=1024,H=16,HD=64 on gfx1250 (MI455X).
//  1) QKV GEMM  : fp32 in -> bf16-WMMA -> bf16 qkv intermediate (halves traffic)
//  2) flash attn: bf16 K-tiles via TDM tensor_load_to_lds (padded LDS layout),
//                 bf16 V staged/transposed, all math v_wmma_f32_16x16x32_bf16
//  3) out GEMM  : bf16 A-tiles via global_load_async_to_lds_b128, fp32 out
// ---------------------------------------------------------------------------

typedef __attribute__((ext_vector_type(8)))  float  v8f;
typedef __attribute__((ext_vector_type(16))) __bf16 v16bf;
typedef __attribute__((ext_vector_type(8)))  __bf16 v8bf;
typedef __attribute__((ext_vector_type(4)))  unsigned int v4u;
typedef __attribute__((ext_vector_type(8)))  int    v8i;
typedef __attribute__((ext_vector_type(4)))  int    v4i;

union ABFrag {
    v16bf  v;
    v8bf   h[2];
    __bf16 e[16];
};

// fp32 -> bf16, round-to-nearest-even
__device__ inline __bf16 tobf(float f) {
    unsigned u = __builtin_bit_cast(unsigned, f);
    u += 0x7FFFu + ((u >> 16) & 1u);
    unsigned short s = (unsigned short)(u >> 16);
    return __builtin_bit_cast(__bf16, s);
}

// LDS byte address of a shared-memory pointer (generic addr low 32 bits are the
// wave-relative LDS offset per ISA 10.2 aperture rules).
__device__ inline unsigned lds_addr_of(const void* p) {
    return (unsigned)(unsigned long long)p;
}

// async 16B global->LDS copy (ASYNCcnt)
__device__ inline void async_cp16(unsigned lds, const void* gaddr) {
    asm volatile("global_load_async_to_lds_b128 %0, %1, off"
                 :: "v"(lds), "v"(gaddr) : "memory");
}
__device__ inline void wait_asynccnt0() {
#if __has_builtin(__builtin_amdgcn_s_wait_asynccnt)
    __builtin_amdgcn_s_wait_asynccnt(0);
#else
    asm volatile("s_wait_asynccnt 0x0" ::: "memory");
#endif
}

#if __has_builtin(__builtin_amdgcn_tensor_load_to_lds)
#define HAVE_TDM 1
#endif

#define LDT 40  // padded LDS row stride (bf16): 80B rows, 16B-aligned chunks
#define LDK 72  // padded K-tile row stride (bf16): 144B rows (TDM pad produces this)

// ---------------------------------------------------------------------------
// C[M,N] = A[M,K] * B[K,N] + bias[N]
// TA in {float, __bf16}; B fp32; TOut in {float, __bf16}.
// 128 thr = 4 waves; 64x64 tile; each wave 2x2 WMMA 16x16 tiles; BK=32.
// ---------------------------------------------------------------------------
template <typename TA, typename TOut>
__global__ __launch_bounds__(128) void gemm_bf16(
    const TA* __restrict__ A, const float* __restrict__ Bm,
    const float* __restrict__ bias, TOut* __restrict__ C,
    int M, int N, int K)
{
    __shared__ __align__(16) __bf16 sA[64 * LDT];   // [m][k]
    __shared__ __align__(16) __bf16 sB[64 * LDT];   // transposed: [n][k]

    const int t    = threadIdx.x;
    const int lane = t & 31;
    const int wid  = t >> 5;
    const int lh   = lane & 15;
    const int sel  = lane >> 4;
    const int wm   = (wid >> 1) * 32;
    const int wn   = (wid & 1) * 32;
    const int m0   = blockIdx.y * 64;
    const int n0   = blockIdx.x * 64;

    v8f acc[2][2] = {};

    for (int k0 = 0; k0 < K; k0 += 32) {
        __syncthreads();
        if constexpr (sizeof(TA) == 4) {
            // fp32 A: load + convert (coalesced along k)
            #pragma unroll
            for (int i = 0; i < 16; ++i) {
                int l = i * 128 + t;
                int r = l >> 5, c = l & 31;
                sA[r * LDT + c] = tobf((float)A[(size_t)(m0 + r) * K + k0 + c]);
            }
        } else {
            // bf16 A: async DMA straight into LDS, 16B per chunk (ASYNCcnt)
            #pragma unroll
            for (int i = 0; i < 2; ++i) {
                int ch = i * 128 + t;          // 256 chunks = 64 rows x 4
                int r = ch >> 2, hc = ch & 3;
                async_cp16(lds_addr_of(&sA[r * LDT + hc * 8]),
                           &A[(size_t)(m0 + r) * K + k0 + hc * 8]);
            }
            wait_asynccnt0();
        }
        // fp32 B tile 32x64, stored transposed (coalesced along n)
        #pragma unroll
        for (int i = 0; i < 16; ++i) {
            int l = i * 128 + t;
            int kk = l >> 6, n = l & 63;
            sB[n * LDT + kk] = tobf(Bm[(size_t)(k0 + kk) * N + n0 + n]);
        }
        __syncthreads();

        ABFrag a[2], b[2];
        #pragma unroll
        for (int mt = 0; mt < 2; ++mt) {
            int row = wm + mt * 16 + lh;
            a[mt].h[0] = *(const v8bf*)&sA[row * LDT + sel * 8];
            a[mt].h[1] = *(const v8bf*)&sA[row * LDT + sel * 8 + 16];
        }
        #pragma unroll
        for (int nt = 0; nt < 2; ++nt) {
            int col = wn + nt * 16 + lh;
            b[nt].h[0] = *(const v8bf*)&sB[col * LDT + sel * 8];
            b[nt].h[1] = *(const v8bf*)&sB[col * LDT + sel * 8 + 16];
        }
        #pragma unroll
        for (int mt = 0; mt < 2; ++mt)
            #pragma unroll
            for (int nt = 0; nt < 2; ++nt)
                acc[mt][nt] = __builtin_amdgcn_wmma_f32_16x16x32_bf16(
                    false, a[mt].v, false, b[nt].v, (short)0, acc[mt][nt],
                    false, false);
    }

    #pragma unroll
    for (int mt = 0; mt < 2; ++mt)
        #pragma unroll
        for (int nt = 0; nt < 2; ++nt) {
            int col = n0 + wn + nt * 16 + lh;
            float bv = bias[col];
            #pragma unroll
            for (int r = 0; r < 8; ++r) {
                int row = m0 + wm + mt * 16 + r + sel * 8;
                float v = acc[mt][nt][r] + bv;
                if constexpr (sizeof(TOut) == 4)
                    C[(size_t)row * N + col] = v;
                else
                    C[(size_t)row * N + col] = tobf(v);
            }
        }
}

// ---------------------------------------------------------------------------
// Flash attention over bf16 qkv[B*S, 3D] (head h cols: q=h*192, k=+64, v=+128).
// Grid (S/64, H, B); 128 thr = 4 waves; wave = 16 q rows; 32 keys/iter.
// K-tile fetched by the Tensor Data Mover with LDS padding (stride 72 halves).
// Output stored with the reference's scrambled headless reshape.
// ---------------------------------------------------------------------------
__global__ __launch_bounds__(128) void flash_attn(
    const __bf16* __restrict__ qkv, const float* __restrict__ mask,
    __bf16* __restrict__ vals)
{
    __shared__ __align__(16) __bf16 sK[32 * LDK];     // [key][hd]
    __shared__ __align__(16) __bf16 sVt[64 * LDT];    // [hd][key]
    __shared__ __align__(16) __bf16 sP[4][16 * LDT];  // per-wave P re-layout

    const int t    = threadIdx.x;
    const int lane = t & 31;
    const int wid  = t >> 5;
    const int lh   = lane & 15;
    const int sel  = lane >> 4;
    const int b    = blockIdx.z;
    const int h    = blockIdx.y;
    const int q0   = blockIdx.x * 64 + wid * 16;

    const size_t rs = 3072;
    const __bf16* base = qkv + (size_t)b * 2048 * rs + h * 192;

    // ---- Q fragments: direct 16B bf16 loads, no conversion ----
    ABFrag qa[2];
    {
        const __bf16* qr = base + (size_t)(q0 + lh) * rs;
        #pragma unroll
        for (int c = 0; c < 2; ++c) {
            qa[c].h[0] = *(const v8bf*)(qr + c * 32 + sel * 8);
            qa[c].h[1] = *(const v8bf*)(qr + c * 32 + sel * 8 + 16);
        }
    }

    float mrow[8], lrow[8];
    v8f ao[4] = {};
    #pragma unroll
    for (int r = 0; r < 8; ++r) { mrow[r] = -1e30f; lrow[r] = 0.0f; }

    const float scale = 0.125f;

    for (int jb = 0; jb < 2048; jb += 32) {
        __syncthreads();
        const __bf16* kbase = base + (size_t)jb * rs + 64;   // K chunk start
        const __bf16* vbase = base + (size_t)jb * rs + 128;  // V chunk start

#if HAVE_TDM
        // ---- K tile 32x64 bf16 via Tensor Data Mover, pad 4 DW / 32 DW ----
        if (wid == 0) {
            unsigned long long ga = (unsigned long long)kbase;
            v4u g0;
            g0[0] = 1u;                                  // count=1, load, user
            g0[1] = lds_addr_of(&sK[0]);                 // lds_addr
            g0[2] = (unsigned)ga;                        // global_addr[31:0]
            g0[3] = (unsigned)((ga >> 32) & 0x01FFFFFFull) | (2u << 30); // type=2
            v8i g1;
            g1[0] = (int)((1u << 16)       // data_size = 2B
                        | (1u << 20)       // pad_enable
                        | (4u << 22)       // pad_interval: 32 DWORDs
                        | (3u << 25));     // pad_amount:   4 DWORDs -> stride 72
            g1[1] = (int)(64u << 16);      // tensor_dim0 = 64 (bits 79:48 lo)
            g1[2] = (int)(32u << 16);      // tensor_dim1 = 32 (bits 111:80 lo)
            g1[3] = (int)(64u << 16);      // tile_dim0   = 64 (bits 127:112)
            g1[4] = 32;                    // tile_dim1   = 32
            g1[5] = 3072;                  // tensor_dim0_stride (elems)
            g1[6] = 0;
            g1[7] = 0;
            v4i z4 = {};
            v8i z8 = {};
            __builtin_amdgcn_tensor_load_to_lds(g0, g1, z4, z4, z8, 0);
            __builtin_amdgcn_s_wait_tensorcnt((unsigned short)0);
        }
#else
        // ---- K tile via async global->LDS (ASYNCcnt) ----
        #pragma unroll
        for (int i = 0; i < 2; ++i) {
            int ch = i * 128 + t;          // 256 chunks = 32 keys x 8
            int key = ch >> 3, hc = ch & 7;
            async_cp16(lds_addr_of(&sK[key * LDK + hc * 8]),
                       kbase + (size_t)key * rs + hc * 8);
        }
        wait_asynccnt0();
#endif
        // ---- V tile 32x64 -> transposed [hd][key] (needs VGPR pass) ----
        #pragma unroll
        for (int i = 0; i < 2; ++i) {
            int ch = i * 128 + t;          // 256 chunks = 32 keys x 8
            int key = ch >> 3, hc = ch & 7;
            v8bf tv = *(const v8bf*)(vbase + (size_t)key * rs + hc * 8);
            #pragma unroll
            for (int j = 0; j < 8; ++j)
                sVt[(hc * 8 + j) * LDT + key] = tv[j];
        }
        __syncthreads();

        // ---- scores: two 16(q)x16(key) tiles, contraction over hd=64 ----
        v8f s[2] = {};
        #pragma unroll
        for (int kt = 0; kt < 2; ++kt) {
            #pragma unroll
            for (int c = 0; c < 2; ++c) {
                ABFrag kb;
                const __bf16* kp = &sK[(kt * 16 + lh) * LDK + c * 32 + sel * 8];
                kb.h[0] = *(const v8bf*)kp;
                kb.h[1] = *(const v8bf*)(kp + 16);
                s[kt] = __builtin_amdgcn_wmma_f32_16x16x32_bf16(
                    false, qa[c].v, false, kb.v, (short)0, s[kt], false, false);
            }
        }

        // ---- online softmax (rows r + sel*8; keys across 16 lanes) ----
        float mnew[8];
        #pragma unroll
        for (int r = 0; r < 8; ++r) {
            int q = q0 + r + sel * 8;
            const float* mp = mask + ((size_t)b * 2048 + q) * 2048 + jb;
            float v0 = s[0][r] * scale + mp[lh];
            float v1 = s[1][r] * scale + mp[16 + lh];
            s[0][r] = v0; s[1][r] = v1;
            float mx = fmaxf(v0, v1);
            #pragma unroll
            for (int off = 8; off >= 1; off >>= 1)
                mx = fmaxf(mx, __shfl_xor(mx, off, 16));
            mnew[r] = fmaxf(mrow[r], mx);
        }
        float alpha[8];
        #pragma unroll
        for (int r = 0; r < 8; ++r) {
            alpha[r] = __expf(mrow[r] - mnew[r]);
            float p0 = __expf(s[0][r] - mnew[r]);
            float p1 = __expf(s[1][r] - mnew[r]);
            s[0][r] = p0; s[1][r] = p1;
            float sum = p0 + p1;
            #pragma unroll
            for (int off = 8; off >= 1; off >>= 1)
                sum += __shfl_xor(sum, off, 16);
            lrow[r] = lrow[r] * alpha[r] + sum;
            mrow[r] = mnew[r];
        }
        #pragma unroll
        for (int nt = 0; nt < 4; ++nt)
            #pragma unroll
            for (int r = 0; r < 8; ++r)
                ao[nt][r] *= alpha[r];

        // ---- P: C-layout -> A-layout via per-wave LDS pad ----
        #pragma unroll
        for (int r = 0; r < 8; ++r) {
            int q = r + sel * 8;
            sP[wid][q * LDT + lh]      = tobf(s[0][r]);
            sP[wid][q * LDT + 16 + lh] = tobf(s[1][r]);
        }
        __builtin_amdgcn_wave_barrier();  // same-wave DS ops are in-order

        ABFrag pa;
        pa.h[0] = *(const v8bf*)&sP[wid][lh * LDT + sel * 8];
        pa.h[1] = *(const v8bf*)&sP[wid][lh * LDT + sel * 8 + 16];

        // ---- O += P @ V ----
        #pragma unroll
        for (int nt = 0; nt < 4; ++nt) {
            ABFrag vb;
            const __bf16* vp = &sVt[(nt * 16 + lh) * LDT + sel * 8];
            vb.h[0] = *(const v8bf*)vp;
            vb.h[1] = *(const v8bf*)(vp + 16);
            ao[nt] = __builtin_amdgcn_wmma_f32_16x16x32_bf16(
                false, pa.v, false, vb.v, (short)0, ao[nt], false, false);
        }
    }

    // ---- normalize + scrambled store (faithful headless reshape), bf16 ----
    #pragma unroll
    for (int r = 0; r < 8; ++r) {
        float inv = 1.0f / lrow[r];
        int sidx = q0 + r + sel * 8;
        int row  = h * 128 + (sidx >> 4);
        int cb   = (sidx & 15) << 6;
        #pragma unroll
        for (int nt = 0; nt < 4; ++nt) {
            int d = nt * 16 + lh;
            vals[((size_t)b * 2048 + row) * 1024 + cb + d] = tobf(ao[nt][r] * inv);
        }
    }
}

// ---------------------------------------------------------------------------
extern "C" void kernel_launch(void* const* d_in, const int* in_sizes, int n_in,
                              void* d_out, int out_size, void* d_ws, size_t ws_size,
                              hipStream_t stream) {
    (void)in_sizes; (void)n_in; (void)out_size; (void)ws_size;
    const float* x    = (const float*)d_in[0];
    const float* mask = (const float*)d_in[1];
    const float* Wqkv = (const float*)d_in[2];
    const float* bqkv = (const float*)d_in[3];
    const float* Wo   = (const float*)d_in[4];
    const float* bo   = (const float*)d_in[5];
    float* out = (float*)d_out;

    __bf16* qkvb  = (__bf16*)d_ws;                    // 4096 x 3072 bf16
    __bf16* valsb = qkvb + (size_t)4096 * 3072;       // 4096 x 1024 bf16

    dim3 blk(128);
    gemm_bf16<float, __bf16><<<dim3(3072 / 64, 4096 / 64), blk, 0, stream>>>(
        x, Wqkv, bqkv, qkvb, 4096, 3072, 1024);
    flash_attn<<<dim3(2048 / 64, 16, 2), blk, 0, stream>>>(qkvb, mask, valsb);
    gemm_bf16<__bf16, float><<<dim3(1024 / 64, 4096 / 64), blk, 0, stream>>>(
        valsb, Wo, bo, out, 4096, 1024, 1024);
}
